// CRF_10015863734518
// MI455X (gfx1250) — matrix operand order repaired and checked by reading, not compiled, not run on hardware
//
#include <hip/hip_runtime.h>

typedef __attribute__((ext_vector_type(16))) _Float16 v16h;
typedef __attribute__((ext_vector_type(2)))  __fp16   fp16x2;
typedef __attribute__((ext_vector_type(8)))  float    v8f;
typedef __attribute__((ext_vector_type(4)))  float    f4;

#define TGT 48
#define SEQN 2048
#define KPAD 64
#define START_TAG 46
#define END_TAG 47

static __device__ __forceinline__ unsigned pk16(float a, float b) {
  fp16x2 q = __builtin_amdgcn_cvt_pkrtz(a, b);   // v_cvt_pk_f16_f32 (RTZ)
  return __builtin_bit_cast(unsigned, q);
}

static __device__ __forceinline__ void sched_pin() {
#if __has_builtin(__builtin_amdgcn_sched_barrier)
  __builtin_amdgcn_sched_barrier(0);
#endif
}

// One wave owns 16 batch rows. Exp-domain CRF scan:
//   x_{t+1} = exp(emit_t) .* (E x_t),  E = exp(transitions), rescaled by row max.
// D[tag,batch] = E[48x48] * X^T[48x16]: 3 M-tiles x 2 K-chunks of
// v_wmma_f32_16x16x32_f16 (K padded 48->64, zeros parked in LDS).
// Emissions are software-pipelined one step ahead (loads top, exps bottom).
__global__ __launch_bounds__(32) void crf_fwd_wmma(
    const float* __restrict__ feat,    // [256, 2048, 48]
    const float* __restrict__ trans,   // [48, 48]  trans[next, prev]
    const int*   __restrict__ slen,    // [256]
    float*       __restrict__ out)     // [256]
{
  __shared__ __align__(32) _Float16 xb[16 * KPAD];  // [batch][k] f16, row = 128B
  __shared__ __align__(16) float    fin[16 * TGT];  // captured scaled x at seq_len

  const int lane = threadIdx.x;
  const int b    = lane & 15;        // batch-in-tile (N-lane / A M-lane)
  const int h    = lane >> 4;        // half-wave
  const int gb   = blockIdx.x * 16 + b;

  // ---- init x0: zeros except x[START_TAG] = 1
  if (lane < 16) {
    uint4* row = (uint4*)&xb[b * KPAD];
    uint4  z;  z.x = 0u; z.y = 0u; z.z = 0u; z.w = 0u;
#pragma unroll
    for (int i = 0; i < 8; ++i) row[i] = z;        // zero 128B incl. K-pad 48..63
    xb[b * KPAD + START_TAG] = (_Float16)1.0f;
  }
  __syncthreads();

  // ---- preload E = exp(trans) as A-fragments (M=tag, K=prev)
  v16h Efrag[3][2];
#pragma unroll
  for (int t = 0; t < 3; ++t) {
    const int m = 16 * t + b;
#pragma unroll
    for (int c = 0; c < 2; ++c) {
      v16h f;
#pragma unroll
      for (int i = 0; i < 16; ++i) {
        const int kl = 16 * h + i;
        const int k  = 32 * c + kl;
        float v = (k < TGT) ? __expf(trans[m * TGT + k]) : 0.0f;
        f[i] = (_Float16)v;
      }
      Efrag[t][c] = f;
    }
  }

  // ---- per-lane length, wave-max length
  const int mylen = slen[gb];
  int L = mylen, o;
  o = __builtin_amdgcn_ds_swizzle(L, (1 << 10) | 0x1f);  L = L > o ? L : o;
  o = __builtin_amdgcn_ds_swizzle(L, (2 << 10) | 0x1f);  L = L > o ? L : o;
  o = __builtin_amdgcn_ds_swizzle(L, (4 << 10) | 0x1f);  L = L > o ? L : o;
  o = __builtin_amdgcn_ds_swizzle(L, (8 << 10) | 0x1f);  L = L > o ? L : o;
  o = __builtin_amdgcn_ds_swizzle(L, (16 << 10) | 0x1f); L = L > o ? L : o;
  L = __builtin_amdgcn_readfirstlane(L);

  const float* ep = feat + (size_t)gb * SEQN * TGT + 8 * h;

  v8f zero8 = {};
  float off = 0.0f, off_fin = 0.0f;

  // ---- prologue: emissions for step 0
  float ee[24];
  {
    f4 c0 = *(const f4*)(ep + 0);
    f4 c1 = *(const f4*)(ep + 4);
    f4 c2 = *(const f4*)(ep + 16);
    f4 c3 = *(const f4*)(ep + 20);
    f4 c4 = *(const f4*)(ep + 32);
    f4 c5 = *(const f4*)(ep + 36);
#pragma unroll
    for (int r = 0; r < 4; ++r) {
      ee[r]      = __expf(c0[r]);  ee[4 + r]  = __expf(c1[r]);
      ee[8 + r]  = __expf(c2[r]);  ee[12 + r] = __expf(c3[r]);
      ee[16 + r] = __expf(c4[r]);  ee[20 + r] = __expf(c5[r]);
    }
  }

  for (int s = 0; s < L; ++s) {
    // ---- prefetch next step's emissions (distance-1 pipeline)
    const int sn = (s + 1 < L) ? (s + 1) : s;
    const float* es = ep + (size_t)sn * TGT;
    f4 n0 = *(const f4*)(es + 0);
    f4 n1 = *(const f4*)(es + 4);
    f4 n2 = *(const f4*)(es + 16);
    f4 n3 = *(const f4*)(es + 20);
    f4 n4 = *(const f4*)(es + 32);
    f4 n5 = *(const f4*)(es + 36);
    sched_pin();  // keep the loads issued here, not sunk to their use

    // ---- B-fragments: lane(b) reads xb[b][32c + 16h .. +15]
    const _Float16* xrow = &xb[b * KPAD + 16 * h];
    v16h B0 = *(const v16h*)(xrow);
    v16h B1 = *(const v16h*)(xrow + 32);

    // ---- y = E * x
    v8f a0 = __builtin_amdgcn_wmma_f32_16x16x32_f16(false, Efrag[0][0], false, B0, (short)0, zero8, false, false);
    a0     = __builtin_amdgcn_wmma_f32_16x16x32_f16(false, Efrag[0][1], false, B1, (short)0, a0,    false, false);
    v8f a1 = __builtin_amdgcn_wmma_f32_16x16x32_f16(false, Efrag[1][0], false, B0, (short)0, zero8, false, false);
    a1     = __builtin_amdgcn_wmma_f32_16x16x32_f16(false, Efrag[1][1], false, B1, (short)0, a1,    false, false);
    v8f a2 = __builtin_amdgcn_wmma_f32_16x16x32_f16(false, Efrag[2][0], false, B0, (short)0, zero8, false, false);
    a2     = __builtin_amdgcn_wmma_f32_16x16x32_f16(false, Efrag[2][1], false, B1, (short)0, a2,    false, false);

    // ---- x = y .* exp(emit)   (ee precomputed last iteration)
    float x[24];
#pragma unroll
    for (int r = 0; r < 8; ++r) x[r]      = a0[r] * ee[r];
#pragma unroll
    for (int r = 0; r < 8; ++r) x[8 + r]  = a1[r] * ee[8 + r];
#pragma unroll
    for (int r = 0; r < 8; ++r) x[16 + r] = a2[r] * ee[16 + r];

    // ---- per-batch max over 48 tags: max3 tree + half-wave swap
    float t0 = fmaxf(fmaxf(x[0],  x[1]),  x[2]);
    float t1 = fmaxf(fmaxf(x[3],  x[4]),  x[5]);
    float t2 = fmaxf(fmaxf(x[6],  x[7]),  x[8]);
    float t3 = fmaxf(fmaxf(x[9],  x[10]), x[11]);
    float t4 = fmaxf(fmaxf(x[12], x[13]), x[14]);
    float t5 = fmaxf(fmaxf(x[15], x[16]), x[17]);
    float t6 = fmaxf(fmaxf(x[18], x[19]), x[20]);
    float t7 = fmaxf(fmaxf(x[21], x[22]), x[23]);
    float u0 = fmaxf(fmaxf(t0, t1), t2);
    float u1 = fmaxf(fmaxf(t3, t4), t5);
    float m  = fmaxf(fmaxf(u0, u1), fmaxf(t6, t7));
    float mo = __uint_as_float((unsigned)__builtin_amdgcn_ds_swizzle(
        (int)__float_as_uint(m), 0x401F /* SWAPX16 */));
    m = fmaxf(m, mo);
    const float rm = __builtin_amdgcn_rcpf(m);
    off += __logf(m);   // off the critical path (feeds only `off`)

    // ---- rescale, pack f16 pairs, write x̂ back to LDS
    float xs[24];
#pragma unroll
    for (int r = 0; r < 24; ++r) xs[r] = x[r] * rm;
#pragma unroll
    for (int t = 0; t < 3; ++t) {
      uint4 st;
      st.x = pk16(xs[8 * t + 0], xs[8 * t + 1]);
      st.y = pk16(xs[8 * t + 2], xs[8 * t + 3]);
      st.z = pk16(xs[8 * t + 4], xs[8 * t + 5]);
      st.w = pk16(xs[8 * t + 6], xs[8 * t + 7]);
      *(uint4*)(&xb[b * KPAD + 16 * t + 8 * h]) = st;
    }

    // ---- capture finals at this batch's seq_len (rare, exec-masked)
    if (s + 1 == mylen) {
#pragma unroll
      for (int t = 0; t < 3; ++t)
#pragma unroll
        for (int r = 0; r < 8; ++r)
          fin[b * TGT + 16 * t + 8 * h + r] = xs[8 * t + r];
      off_fin = off;
    }

    // ---- exps for next step (fills WMMA/DS latency; loads landed long ago)
#pragma unroll
    for (int r = 0; r < 4; ++r) {
      ee[r]      = __expf(n0[r]);  ee[4 + r]  = __expf(n1[r]);
      ee[8 + r]  = __expf(n2[r]);  ee[12 + r] = __expf(n3[r]);
      ee[16 + r] = __expf(n4[r]);  ee[20 + r] = __expf(n5[r]);
    }
  }
  __syncthreads();

  // ---- terminal: score = off_fin + log( sum_n fin[b][n] * exp(trans[END, n]) )
  float sum = 0.0f;
#pragma unroll
  for (int t = 0; t < 3; ++t)
#pragma unroll
    for (int r = 0; r < 8; ++r) {
      const int mt = 16 * t + 8 * h + r;
      sum += __expf(trans[END_TAG * TGT + mt]) * fin[b * TGT + mt];
    }
  float so = __uint_as_float((unsigned)__builtin_amdgcn_ds_swizzle(
      (int)__float_as_uint(sum), 0x401F));
  sum += so;
  if (lane < 16) out[blockIdx.x * 16 + b] = off_fin + __logf(sum);
}

extern "C" void kernel_launch(void* const* d_in, const int* in_sizes, int n_in,
                              void* d_out, int out_size, void* d_ws, size_t ws_size,
                              hipStream_t stream) {
  (void)in_sizes; (void)n_in; (void)out_size; (void)d_ws; (void)ws_size;
  const float* feat  = (const float*)d_in[0];
  const float* trans = (const float*)d_in[1];
  const int*   slen  = (const int*)d_in[2];
  float*       out   = (float*)d_out;
  crf_fwd_wmma<<<dim3(16), dim3(32), 0, stream>>>(feat, trans, slen, out);
}
